// SAGENet_60816736911616
// MI455X (gfx1250) — matrix operand order repaired and checked by reading, not compiled, not run on hardware
//
#include <hip/hip_runtime.h>
#include <hip/hip_bf16.h>

// ---------------------------------------------------------------------------
// Types for WMMA (gfx1250, wave32)
// ---------------------------------------------------------------------------
typedef __attribute__((ext_vector_type(16))) __bf16          v16bf;
typedef __attribute__((ext_vector_type(8)))  float           v8f;
typedef __attribute__((ext_vector_type(8)))  unsigned short  v8us;
typedef __attribute__((ext_vector_type(16))) unsigned short  v16us;

#define D_FEAT 128   // D_IN == D_HID == 128
#define N_CLS  40
#define N_PAD  48    // 40 padded to 3 tiles of 16

// float -> bf16, round to nearest even
__device__ __forceinline__ unsigned short f2bf(float f) {
    unsigned int u = __float_as_uint(f);
    unsigned int r = u + 0x7FFFu + ((u >> 16) & 1u);
    return (unsigned short)(r >> 16);
}

union FragCvt { struct { v8us lo, hi; } p; v16bf v; };

// A fragment (16x32 bf16, M x K): per-lane elements are two contiguous runs
// of 8: K = base..base+7 (elems 0-7) and base+16..base+23 (elems 8-15),
// where base = k0 + (lane>=16 ? 8 : 0). `p` points at row start + base.
__device__ __forceinline__ v16bf load_a_frag(const unsigned short* p) {
    FragCvt c;
    c.p.lo = *(const v8us*)(p);
    c.p.hi = *(const v8us*)(p + 16);
    return c.v;
}

// B fragment (32x16 bf16, K x N), weights stored column-major (WT[n*128+k]):
// per-lane two runs of 8: K = base..base+7 and base+16..base+23, base = k0 + (lane>=16 ? 8 : 0).
__device__ __forceinline__ v16bf load_b_frag(const unsigned short* p) {
    FragCvt c;
    c.p.lo = *(const v8us*)(p);
    c.p.hi = *(const v8us*)(p + 16);
    return c.v;
}

// ---------------------------------------------------------------------------
// Utility kernels
// ---------------------------------------------------------------------------
__global__ void zero_f32(float* __restrict__ p, size_t n) {
    size_t i = (size_t)blockIdx.x * blockDim.x + threadIdx.x;
    if (i < n) p[i] = 0.0f;
}

__global__ void cvt_f32_bf16(const float* __restrict__ src,
                             unsigned short* __restrict__ dst, size_t n) {
    size_t i = (size_t)blockIdx.x * blockDim.x + threadIdx.x;
    if (i < n) dst[i] = f2bf(src[i]);
}

// Transpose W [128 x C] (row-major, K-major) into WT [PC x 128] column-major
// bf16, zero-padding columns C..PC-1.
__global__ void cvt_weight_t(const float* __restrict__ W,
                             unsigned short* __restrict__ WT, int C, int PC) {
    int i = blockIdx.x * blockDim.x + threadIdx.x;
    if (i >= PC * D_FEAT) return;
    int n = i >> 7;          // padded output column
    int k = i & (D_FEAT - 1);
    WT[(size_t)n * D_FEAT + k] = (n < C) ? f2bf(W[(size_t)k * C + n])
                                         : (unsigned short)0;
}

__global__ void deg_kernel(const int* __restrict__ dst, float* __restrict__ deg,
                           int E) {
    int t = blockIdx.x * blockDim.x + threadIdx.x;
    if (t < E) atomicAdd(&deg[dst[t]], 1.0f);
}

// One thread per (edge, 4-feature chunk): 32 threads cover an edge's 128 feats.
__global__ void scatter_add(const float* __restrict__ feat,
                            const int* __restrict__ src,
                            const int* __restrict__ dst,
                            float* __restrict__ agg, int E) {
    int t = blockIdx.x * blockDim.x + threadIdx.x;
    int e = t >> 5;
    if (e >= E) return;
    int c = (t & 31) << 2;
    int s = src[e], d = dst[e];
    const float4 f = *(const float4*)(feat + (size_t)s * D_FEAT + c);
    float* p = agg + (size_t)d * D_FEAT + c;
    atomicAdd(p + 0, f.x);
    atomicAdd(p + 1, f.y);
    atomicAdd(p + 2, f.z);
    atomicAdd(p + 3, f.w);
}

// mean = agg / max(deg,1), converted to bf16
__global__ void mean_to_bf16(const float* __restrict__ agg,
                             const float* __restrict__ deg,
                             unsigned short* __restrict__ mb, size_t total) {
    size_t i = (size_t)blockIdx.x * blockDim.x + threadIdx.x;
    if (i >= total) return;
    float d = deg[i >> 7];
    mb[i] = f2bf(agg[i] / fmaxf(d, 1.0f));
}

// ---------------------------------------------------------------------------
// Fused SAGE GEMM:  out = A1 @ Wl + A2 @ Wr + bias  (+ optional ReLU)
// A1/A2: [Nnodes x 128] bf16; Wl/Wr: col-major bf16 [ncols x 128].
// One wave computes one 16x16 output tile; 8 waves per block.
// ---------------------------------------------------------------------------
__global__ __launch_bounds__(256)
void sage_gemm_bf16(const unsigned short* __restrict__ A1,
                    const unsigned short* __restrict__ A2,
                    const unsigned short* __restrict__ Bl,
                    const unsigned short* __restrict__ Br,
                    const float* __restrict__ bias, int biasN,
                    float* __restrict__ outF,
                    unsigned short* __restrict__ outB,
                    int ldo, int tilesN, int totalTiles, int relu) {
    int tile = blockIdx.x * 8 + threadIdx.y;
    if (tile >= totalTiles) return;          // uniform per wave
    int tm = tile / tilesN;
    int tn = tile - tm * tilesN;
    int m0 = tm << 4, n0 = tn << 4;

    int lane = threadIdx.x;
    int hi = lane >> 4;        // which half-wave
    int lr = lane & 15;

    const unsigned short* a1 = A1 + (size_t)(m0 + lr) * D_FEAT + hi * 8;
    const unsigned short* a2 = A2 + (size_t)(m0 + lr) * D_FEAT + hi * 8;
    const unsigned short* bl = Bl + (size_t)(n0 + lr) * D_FEAT + hi * 8;
    const unsigned short* br = Br + (size_t)(n0 + lr) * D_FEAT + hi * 8;

    v8f acc = {};
#pragma unroll
    for (int k0 = 0; k0 < D_FEAT; k0 += 32) {
        v16bf afl = load_a_frag(a1 + k0);
        v16bf bfl = load_b_frag(bl + k0);
        acc = __builtin_amdgcn_wmma_f32_16x16x32_bf16(
            false, afl, false, bfl, (short)0, acc, false, false);
        v16bf afr = load_a_frag(a2 + k0);
        v16bf bfr = load_b_frag(br + k0);
        acc = __builtin_amdgcn_wmma_f32_16x16x32_bf16(
            false, afr, false, bfr, (short)0, acc, false, false);
    }

    int col = n0 + lr;
    float bv = (col < biasN) ? bias[col] : 0.0f;
#pragma unroll
    for (int v = 0; v < 8; ++v) {
        int row = m0 + hi * 8 + v;
        float val = acc[v] + bv;
        if (relu) val = fmaxf(val, 0.0f);
        size_t o = (size_t)row * ldo + col;
        outF[o] = val;
        if (outB) outB[o] = f2bf(val);
    }
}

// ---------------------------------------------------------------------------
// Row softmax over 40 classes (input padded to 48 cols). One wave per row.
// ---------------------------------------------------------------------------
__global__ __launch_bounds__(256)
void softmax40(const float* __restrict__ inp, float* __restrict__ out,
               int nrows) {
    int row = blockIdx.x * 8 + threadIdx.y;
    if (row >= nrows) return;
    int l = threadIdx.x;
    const float* r = inp + (size_t)row * N_PAD;
    float v0 = r[l];                                  // cols 0..31
    float v1 = (l < 8) ? r[32 + l] : -3.4e38f;        // cols 32..39
    float m = fmaxf(v0, v1);
#pragma unroll
    for (int off = 16; off > 0; off >>= 1) m = fmaxf(m, __shfl_xor(m, off, 32));
    float e0 = __expf(v0 - m);
    float e1 = (l < 8) ? __expf(v1 - m) : 0.0f;
    float s = e0 + e1;
#pragma unroll
    for (int off = 16; off > 0; off >>= 1) s += __shfl_xor(s, off, 32);
    float inv = 1.0f / s;
    out[(size_t)row * N_CLS + l] = e0 * inv;
    if (l < 8) out[(size_t)row * N_CLS + 32 + l] = e1 * inv;
}

// ---------------------------------------------------------------------------
// Host launcher
// ---------------------------------------------------------------------------
extern "C" void kernel_launch(void* const* d_in, const int* in_sizes, int n_in,
                              void* d_out, int out_size, void* d_ws,
                              size_t ws_size, hipStream_t stream) {
    const float* x    = (const float*)d_in[0];
    const int*   ei   = (const int*)  d_in[1];
    const float* W1l  = (const float*)d_in[2];
    const float* W1r  = (const float*)d_in[3];
    const float* b1   = (const float*)d_in[4];
    const float* W2l  = (const float*)d_in[5];
    const float* W2r  = (const float*)d_in[6];
    const float* b2   = (const float*)d_in[7];
    float* out = (float*)d_out;

    const int N = in_sizes[0] / D_FEAT;   // 100000
    const int E = in_sizes[1] / 2;        // 1600000
    const int* src = ei;
    const int* dst = ei + E;

    // ---- workspace carve-up (256B aligned) ----
    char* w = (char*)d_ws;
    auto carve = [&](size_t bytes) {
        void* p = (void*)w;
        w += (bytes + 255) & ~(size_t)255;
        return p;
    };
    const size_t NF = (size_t)N * D_FEAT;
    float*          agg    = (float*)         carve(NF * 4);          // 51.2MB
    float*          deg    = (float*)         carve((size_t)N * 4);
    unsigned short* meanbf = (unsigned short*)carve(NF * 2);          // 25.6MB
    unsigned short* xbf    = (unsigned short*)carve(NF * 2);          // 25.6MB
    float*          h      = (float*)         carve(NF * 4);          // 51.2MB
    unsigned short* hbf    = (unsigned short*)carve(NF * 2);          // 25.6MB
    unsigned short* w1lt   = (unsigned short*)carve((size_t)D_FEAT * D_FEAT * 2);
    unsigned short* w1rt   = (unsigned short*)carve((size_t)D_FEAT * D_FEAT * 2);
    unsigned short* w2lt   = (unsigned short*)carve((size_t)N_PAD * D_FEAT * 2);
    unsigned short* w2rt   = (unsigned short*)carve((size_t)N_PAD * D_FEAT * 2);
    float*          outp   = agg;  // padded logits alias agg (free by then)

    const int T = 256;
    const int nfBlocks   = (int)((NF + T - 1) / T);
    const int edgeBlocks = (E + T - 1) / T;
    const int scatBlocks = (int)(((size_t)E * 32 + T - 1) / T);

    // ---- weight / input conversion ----
    cvt_weight_t<<<(D_FEAT * D_FEAT + T - 1) / T, T, 0, stream>>>(W1l, w1lt, D_FEAT, D_FEAT);
    cvt_weight_t<<<(D_FEAT * D_FEAT + T - 1) / T, T, 0, stream>>>(W1r, w1rt, D_FEAT, D_FEAT);
    cvt_weight_t<<<(N_PAD * D_FEAT + T - 1) / T, T, 0, stream>>>(W2l, w2lt, N_CLS, N_PAD);
    cvt_weight_t<<<(N_PAD * D_FEAT + T - 1) / T, T, 0, stream>>>(W2r, w2rt, N_CLS, N_PAD);
    cvt_f32_bf16<<<nfBlocks, T, 0, stream>>>(x, xbf, NF);

    // ---- degree (shared by both layers) ----
    zero_f32<<<(N + T - 1) / T, T, 0, stream>>>(deg, (size_t)N);
    deg_kernel<<<edgeBlocks, T, 0, stream>>>(dst, deg, E);

    // ================= layer 1 =================
    zero_f32<<<nfBlocks, T, 0, stream>>>(agg, NF);
    scatter_add<<<scatBlocks, T, 0, stream>>>(x, src, dst, agg, E);
    mean_to_bf16<<<nfBlocks, T, 0, stream>>>(agg, deg, meanbf, NF);

    {
        int tilesM = N / 16, tilesN = D_FEAT / 16;
        int total = tilesM * tilesN;
        dim3 blk(32, 8);
        sage_gemm_bf16<<<(total + 7) / 8, blk, 0, stream>>>(
            meanbf, xbf, w1lt, w1rt, b1, D_FEAT,
            h, hbf, D_FEAT, tilesN, total, /*relu=*/1);
    }

    // ================= layer 2 =================
    zero_f32<<<nfBlocks, T, 0, stream>>>(agg, NF);
    scatter_add<<<scatBlocks, T, 0, stream>>>(h, src, dst, agg, E);
    mean_to_bf16<<<nfBlocks, T, 0, stream>>>(agg, deg, meanbf, NF);

    {
        int tilesM = N / 16, tilesN = N_PAD / 16;
        int total = tilesM * tilesN;
        dim3 blk(32, 8);
        sage_gemm_bf16<<<(total + 7) / 8, blk, 0, stream>>>(
            meanbf, hbf, w2lt, w2rt, b2, N_CLS,
            outp, (unsigned short*)nullptr, N_PAD, tilesN, total, /*relu=*/0);
    }

    // ---- softmax over 40 classes ----
    {
        dim3 blk(32, 8);
        softmax40<<<(N + 7) / 8, blk, 0, stream>>>(outp, out, N);
    }
}
